// NeuralGeneratedConv_53644141527175
// MI455X (gfx1250) — matrix operand order repaired and checked by reading, not compiled, not run on hardware
//
#include <hip/hip_runtime.h>

typedef __attribute__((ext_vector_type(16))) _Float16 v16h;
typedef __attribute__((ext_vector_type(8)))  float    v8f;
typedef int v4i __attribute__((vector_size(16)));
typedef __attribute__((address_space(1))) v4i* as1_v4i_p;
typedef __attribute__((address_space(3))) v4i* as3_v4i_p;

#define CI    16
#define CO    16
#define KK    9          // 3x3 taps
#define CIKK  144        // CI*KK, K-dim of per-c GEMM
#define HID   256
#define CBLK  257        // 256 hidden blocks + 1 bias block
#define NKC   5          // ceil(144/32) K-chunks (padded to 160)
#define IMG_H 128
#define IMG_W 128
#define PIX   16384      // H*W
#define NWAVE 8          // waves per block
#define BLK_HALF (NKC * 32 * 16)     // halves per c-block fragment set (2560)
#define BLK_BYTES (BLK_HALF * 2)     // 5120 bytes per c-block

#if __has_builtin(__builtin_amdgcn_global_load_async_to_lds_b128)
#define USE_ASYNC_LDS 1
#else
#define USE_ASYNC_LDS 0
#endif

__device__ __forceinline__ int reflect_idx(int t, int n) {
    if (t < 0) t = -t;
    if (t >= n) t = 2 * n - 2 - t;
    return t;
}

// A-fragment (16x32 f16) half-index -> K offset within a 32-wide chunk
// (ISA 7.12.2: VGPR0 K=0,1 / lanes16-31 K=8,9; VGPR4-7 K=16..31)
__device__ __forceinline__ int a_frag_k(int e, int g) {
    int v = e >> 1;
    return 16 * (v >> 2) + 2 * (v & 3) + (e & 1) + 8 * g;
}

__device__ __forceinline__ void wait_async_lds() {
#if USE_ASYNC_LDS
#if __has_builtin(__builtin_amdgcn_s_wait_asynccnt)
    __builtin_amdgcn_s_wait_asynccnt(0);
#else
    asm volatile("s_wait_asynccnt 0" ::: "memory");
#endif
#endif
}

// Cooperative async copy of `nblk` consecutive 5120B c-blocks into LDS.
__device__ __forceinline__ void stage_blocks(const _Float16* __restrict__ w2s,
                                             int firstBlk, int nblk,
                                             _Float16* dst, int tid) {
    const int n16 = nblk * (BLK_BYTES / 16);
#if USE_ASYNC_LDS
    as1_v4i_p g1 = (as1_v4i_p)(void*)((const char*)w2s + (size_t)firstBlk * BLK_BYTES);
    as3_v4i_p l3 = (as3_v4i_p)dst;
    for (int idx = tid; idx < n16; idx += 256)
        __builtin_amdgcn_global_load_async_to_lds_b128(g1 + idx, l3 + idx, 0, 0);
#else
    const uint4* gsrc = (const uint4*)((const char*)w2s + (size_t)firstBlk * BLK_BYTES);
    uint4* d = (uint4*)dst;
    for (int idx = tid; idx < n16; idx += 256) d[idx] = gsrc[idx];
#endif
}

// ---------------------------------------------------------------------------
// Prep: convert W2 (f32 [256][2304]) + b2 into pre-swizzled f16 A-fragments.
// Fragment (c, kk, lane) holds A[o = lane%16, K = 32*kk + a_frag_k(e, lane/16)]
// where A[o, iq] = W2[c, o*144 + iq]  (c==256 -> b2), zero-padded for iq>=144.
// ---------------------------------------------------------------------------
__global__ void prep_w2(const float* __restrict__ W2, const float* __restrict__ b2,
                        _Float16* __restrict__ w2s) {
    int idx = blockIdx.x * blockDim.x + threadIdx.x;
    if (idx >= CBLK * NKC * 32) return;
    int l  = idx & 31;
    int kk = (idx >> 5) % NKC;
    int c  = idx / (NKC * 32);
    int o = l & 15, g = l >> 4;
    const float* row = (c < HID) ? (W2 + (size_t)c * (CO * CIKK)) : b2;
    v16h frag;
#pragma unroll
    for (int e = 0; e < 16; ++e) {
        int kidx = 32 * kk + a_frag_k(e, g);
        float v = (kidx < CIKK) ? row[o * CIKK + kidx] : 0.0f;
        frag[e] = (_Float16)v;
    }
    *reinterpret_cast<v16h*>(w2s + (size_t)idx * 16) = frag;
}

// Build B fragments with compile-time lane-group G so all tap indices fold.
// B[K, pixel]: element e <-> K = 32*kk + 8*G + e (e<8), 32*kk + 16 + 8*G + (e-8) (e>=8), column N = lane%16.
template <int G>
__device__ __forceinline__ void build_b(v16h* bf, const float* __restrict__ img,
                                        int y, int x) {
    int ry[3] = { reflect_idx(y - 1, IMG_H), y, reflect_idx(y + 1, IMG_H) };
    int rx[3] = { reflect_idx(x - 1, IMG_W), x, reflect_idx(x + 1, IMG_W) };
#pragma unroll
    for (int kk = 0; kk < NKC; ++kk) {
        v16h t;
#pragma unroll
        for (int e = 0; e < 16; ++e) {
            const int kidx = 32 * kk + (e < 8 ? 8 * G + e : 16 + 8 * G + (e - 8));
            float v = 0.0f;
            if (kidx < CIKK) {
                const int i = kidx / KK, q = kidx % KK;   // compile-time
                v = img[(i << 14) + (ry[q / 3] << 7) + rx[q % 3]];
            }
            t[e] = (_Float16)v;
        }
        bf[kk] = t;
    }
}

// ---------------------------------------------------------------------------
// Heavy kernel: wave = 16 consecutive pixels of one image row.
// Two c-blocks per pipeline stage: double-buffered 10KB LDS pairs, async copy
// of pair i+1 overlapping 2x5 interleaved WMMA chains of pair i; one barrier
// per pair (129 total). Bias block 256 handled in a branch-free epilogue.
// ---------------------------------------------------------------------------
__global__ void __launch_bounds__(256, 1)
hyperconv_wmma(const float* __restrict__ input, const float* __restrict__ foa,
               const float* __restrict__ W1, const float* __restrict__ b1,
               const _Float16* __restrict__ w2s, float* __restrict__ out) {
    __shared__ __align__(32) _Float16 lds[2][2 * BLK_HALF];   // 2 x 10240 B

    const int tid  = threadIdx.x;
    const int lane = tid & 31;
    const int wave = tid >> 5;
    const int g = lane >> 4;        // lane group
    const int n = lane & 15;        // column = local pixel
    const int pixel_base = (blockIdx.x * NWAVE + wave) * 16;
    const int b  = pixel_base >> 14;          // / PIX
    const int p0 = pixel_base & (PIX - 1);
    const int p  = p0 + n;
    const int y = p >> 7, x = p & 127;

    // Patch B-fragments, resident in VGPRs for the whole c-loop.
    const float* img = input + (size_t)b * CI * PIX;
    v16h bf[NKC];
    if (g == 0) build_b<0>(bf, img, y, x);
    else        build_b<1>(bf, img, y, x);

    const float fx = foa[b * 2 + 0], fy = foa[b * 2 + 1];
    const float dxc = (float)x - fx;
    const float dyc = (float)y - fy;

    // Prologue: stage pair {0, 1}.
    stage_blocks(w2s, 0, 2, &lds[0][0], tid);
    wait_async_lds();
    __syncthreads();

    v8f outf = {0.f, 0.f, 0.f, 0.f, 0.f, 0.f, 0.f, 0.f};

    for (int i = 0; i < 128; ++i) {
        const int c = 2 * i;
        // Stage the next pair (last step stages only the bias block 256).
        const int nxt = c + 2;
        stage_blocks(w2s, nxt, (nxt == 256) ? 1 : 2, &lds[(i + 1) & 1][0], tid);

        const _Float16* buf0 = &lds[i & 1][0];
        const _Float16* buf1 = buf0 + BLK_HALF;
        v8f acc0 = {0.f, 0.f, 0.f, 0.f, 0.f, 0.f, 0.f, 0.f};
        v8f acc1 = {0.f, 0.f, 0.f, 0.f, 0.f, 0.f, 0.f, 0.f};
#pragma unroll
        for (int kk = 0; kk < NKC; ++kk) {
            v16h af0 = *reinterpret_cast<const v16h*>(buf0 + (kk * 32 + lane) * 16);
            v16h af1 = *reinterpret_cast<const v16h*>(buf1 + (kk * 32 + lane) * 16);
            acc0 = __builtin_amdgcn_wmma_f32_16x16x32_f16(
                       false, af0, false, bf[kk], (short)0, acc0, false, false);
            acc1 = __builtin_amdgcn_wmma_f32_16x16x32_f16(
                       false, af1, false, bf[kk], (short)0, acc1, false, false);
        }

        float pre0 = fmaf(dxc, W1[c],     fmaf(dyc, W1[HID + c],     b1[c]));
        float pre1 = fmaf(dxc, W1[c + 1], fmaf(dyc, W1[HID + c + 1], b1[c + 1]));
        float hc0 = pre0 > 0.0f ? pre0 : 0.0f;
        float hc1 = pre1 > 0.0f ? pre1 : 0.0f;
#pragma unroll
        for (int r = 0; r < 8; ++r)
            outf[r] += hc0 * acc0[r] + hc1 * acc1[r];

        wait_async_lds();   // next pair resident before publishing
        __syncthreads();
    }

    // Epilogue: bias block (c == 256, h == 1), staged into lds[0] by i == 127.
    {
        const _Float16* buf = &lds[0][0];
        v8f acc = {0.f, 0.f, 0.f, 0.f, 0.f, 0.f, 0.f, 0.f};
#pragma unroll
        for (int kk = 0; kk < NKC; ++kk) {
            v16h af = *reinterpret_cast<const v16h*>(buf + (kk * 32 + lane) * 16);
            acc = __builtin_amdgcn_wmma_f32_16x16x32_f16(
                      false, af, false, bf[kk], (short)0, acc, false, false);
        }
#pragma unroll
        for (int r = 0; r < 8; ++r) outf[r] += acc[r];
    }

    // D fragment: M = o = r + 8*g, N = pixel p
#pragma unroll
    for (int r = 0; r < 8; ++r) {
        int o = r + 8 * g;
        out[((size_t)(b * CO + o) << 14) + p] = outf[r];
    }
}

extern "C" void kernel_launch(void* const* d_in, const int* in_sizes, int n_in,
                              void* d_out, int out_size, void* d_ws, size_t ws_size,
                              hipStream_t stream) {
    const float* input = (const float*)d_in[0];   // [4,16,128,128]
    const float* foa   = (const float*)d_in[1];   // [4,2]
    const float* W1    = (const float*)d_in[2];   // [2,256]
    const float* b1    = (const float*)d_in[3];   // [256]
    const float* W2    = (const float*)d_in[4];   // [256,2304]
    const float* b2    = (const float*)d_in[5];   // [2304]
    float* outp = (float*)d_out;                  // [4,16,128,128]
    _Float16* w2s = (_Float16*)d_ws;              // 257*5*32*16 halves ~ 1.26 MB

    const int prep_n = CBLK * NKC * 32;           // 41120
    prep_w2<<<(prep_n + 255) / 256, 256, 0, stream>>>(W2, b2, w2s);

    // 65536 pixels / (8 waves * 16 pixels) = 512 blocks
    hyperconv_wmma<<<512, NWAVE * 32, 0, stream>>>(input, foa, W1, b1, w2s, outp);
}
